// BinaryTreeGoE_71648644431987
// MI455X (gfx1250) — matrix operand order repaired and checked by reading, not compiled, not run on hardware
//
#include <hip/hip_runtime.h>
#include <hip/hip_bf16.h>

// ---------------------------------------------------------------------------
// BinaryTreeGoE on MI455X (gfx1250): 3 stages of bf16 WMMA GEMM with per-row
// expert routing via atomic compaction (computes only the selected path:
// 3 unit-GEMMs instead of the reference's 7).
//  - v_wmma_f32_16x16x32_bf16 matrix core path, fp32 accumulate
//  - global_load_async_to_lds_b128 (ASYNCcnt) staging for bf16 activations
//  - packed b32 transpose stores for the weight tile
// ---------------------------------------------------------------------------

typedef __attribute__((ext_vector_type(16))) __bf16 v16bf;
typedef __attribute__((ext_vector_type(8)))  __bf16 v8bf;
typedef __attribute__((ext_vector_type(8)))  float  v8f;
typedef __attribute__((ext_vector_type(4)))  float  v4f;

static constexpr int Bsz   = 8192;
static constexpr int Dsz   = 2048;
static constexpr int BM    = 128;
static constexpr int BN    = 128;
static constexpr int BK    = 32;
static constexpr int KT    = Dsz / BK;     // 64 k-tiles
static constexpr int LDA   = BK + 8;       // halves; keeps 16B alignment, spreads banks
static constexpr int LDB   = BK + 8;

__device__ __forceinline__ __bf16 f2bf(float f) {
    union { float f; unsigned u; } v; v.f = f;
    unsigned r = v.u + 0x7fffu + ((v.u >> 16) & 1u);   // round-to-nearest-even
    unsigned short h = (unsigned short)(r >> 16);
    __bf16 b;
    __builtin_memcpy(&b, &h, 2);
    return b;
}

__device__ __forceinline__ unsigned pack_bf2(float lo, float hi) {
    unsigned short a, b;
    __bf16 ba = f2bf(lo), bb = f2bf(hi);
    __builtin_memcpy(&a, &ba, 2);
    __builtin_memcpy(&b, &bb, 2);
    return (unsigned)a | ((unsigned)b << 16);
}

// ---------------------------------------------------------------------------
// Routing: bucket rows by expert for stage1 (2 experts) and stage2 (4 leaves).
// ---------------------------------------------------------------------------
__global__ void zero_counters_kernel(int* c) {
    if (threadIdx.x < 8) c[threadIdx.x] = 0;
}

__global__ void compact_kernel(const int* __restrict__ mask,
                               int* __restrict__ cnt1, int* __restrict__ idx1,
                               int* __restrict__ cnt2, int* __restrict__ idx2) {
    int b = blockIdx.x * blockDim.x + threadIdx.x;
    if (b >= Bsz) return;
    int bit0 = mask[b * 3 + 0] & 1;
    int bit1 = mask[b * 3 + 1] & 1;
    int p1 = atomicAdd(&cnt1[bit0], 1);
    idx1[bit0 * Bsz + p1] = b;
    int e2 = bit0 * 2 + bit1;
    int p2 = atomicAdd(&cnt2[e2], 1);
    idx2[e2 * Bsz + p2] = b;
}

// ---------------------------------------------------------------------------
// GEMM: out[rows(e), :] = relu( A[rows(e), :] @ W[e] + bias[e] )
// A_IS_F32: A operand is fp32 (stage 0 input x) vs bf16 (h0/h1).
// OUT_BF16: store bf16 hidden (stages 0/1) vs fp32 final output (stage 2).
// Grid: (Dsz/BN, Bsz/BM, num_experts); block: 256 threads = 8 wave32.
// ---------------------------------------------------------------------------
template <bool A_IS_F32, bool OUT_BF16>
__global__ __launch_bounds__(256)
void goe_gemm_kernel(const void* __restrict__ Aptr,
                     const float* __restrict__ Wbase,
                     const float* __restrict__ biasBase,
                     void* __restrict__ Outptr,
                     const int* __restrict__ idx,   // nullptr => identity rows
                     const int* __restrict__ cnt) { // nullptr => Bsz rows
    const int e    = blockIdx.z;
    const int rows = idx ? cnt[e] : Bsz;
    const int m0   = blockIdx.y * BM;
    if (m0 >= rows) return;                        // empty tail tile for expert
    const int n0 = blockIdx.x * BN;

    const float* __restrict__ W    = Wbase + (size_t)e * Dsz * Dsz;
    const float* __restrict__ bias = biasBase + (size_t)e * Dsz;
    const int*   __restrict__ rIdx = idx ? (idx + (size_t)e * Bsz) : nullptr;

    __shared__ __bf16 Alds[2][BM * LDA];
    __shared__ __bf16 Blds[2][BN * LDB];

    const int t     = threadIdx.x;
    const int lane  = t & 31;
    const int wave  = t >> 5;
    const int waveM = wave & 3;    // 4 waves over M -> 4*32 = 128
    const int waveN = wave >> 2;   // 2 waves over N -> 2*64 = 128
    const int lr    = lane & 15;
    const int hs    = lane >> 4;   // half-wave select

    // ---- A staging coords: 2 rows x 8 cols (16B) per thread per k-tile ----
    int agrow[2];
    int arow[2];
#pragma unroll
    for (int p = 0; p < 2; ++p) {
        int r   = (t >> 2) + p * 64;   // tile-local row 0..127
        arow[p] = r;
        int gm  = m0 + r;
        agrow[p] = rIdx ? ((gm < rows) ? rIdx[gm] : -1) : gm;
    }
    const int acol = (t & 3) * 8;

    // ---- B staging coords: 2 adjacent k-rows x 8 n-cols per thread ----
    const int bk0  = (t >> 4) * 2;    // 0,2,..,30
    const int bcol = (t & 15) * 8;    // 0..120

    auto stageA = [&](int kt, int buf) {
#pragma unroll
        for (int p = 0; p < 2; ++p) {
            __bf16* dst = &Alds[buf][arow[p] * LDA + acol];
            int gr = agrow[p];
            if (gr >= 0) {
                if constexpr (A_IS_F32) {
                    const float* src = (const float*)Aptr + (size_t)gr * Dsz + kt * BK + acol;
                    v4f x0 = *(const v4f*)(src);
                    v4f x1 = *(const v4f*)(src + 4);
                    v8bf o;
#pragma unroll
                    for (int i = 0; i < 4; ++i) { o[i] = f2bf(x0[i]); o[i + 4] = f2bf(x1[i]); }
                    *(v8bf*)dst = o;
                } else {
                    // bf16 activations: pure 16B copy -> async DMA to LDS
                    // (generic pointer to __shared__ carries the LDS byte
                    //  offset in its low 32 bits per the aperture mapping)
                    const __bf16* src = (const __bf16*)Aptr + (size_t)gr * Dsz + kt * BK + acol;
                    unsigned lds_off = (unsigned)(uintptr_t)dst;
                    asm volatile("global_load_async_to_lds_b128 %0, %1, off"
                                 :: "v"(lds_off), "v"(src)
                                 : "memory");
                }
            } else {
                v8bf z = {};
                *(v8bf*)dst = z;
            }
        }
    };

    auto stageB = [&](int kt, int buf) {
        const float* src0 = W + (size_t)(kt * BK + bk0) * Dsz + n0 + bcol;
        const float* src1 = src0 + Dsz;                         // k+1 row
        __builtin_prefetch(src0 + 2 * BK * Dsz, 0, 0);          // k+2 tile -> L2 hint
        v4f a0 = *(const v4f*)(src0);
        v4f a1 = *(const v4f*)(src0 + 4);
        v4f b0 = *(const v4f*)(src1);
        v4f b1 = *(const v4f*)(src1 + 4);
        // transpose into LDS: Bt[n][k]; adjacent k pair packs into one b32 store
#pragma unroll
        for (int i = 0; i < 8; ++i) {
            float flo = (i < 4) ? a0[i] : a1[i - 4];
            float fhi = (i < 4) ? b0[i] : b1[i - 4];
            *(unsigned*)&Blds[buf][(bcol + i) * LDB + bk0] = pack_bf2(flo, fhi);
        }
    };

    v8f acc[2][4];
#pragma unroll
    for (int fm = 0; fm < 2; ++fm)
#pragma unroll
        for (int fn = 0; fn < 4; ++fn) acc[fm][fn] = (v8f){};

    auto compute = [&](int buf) {
#pragma unroll
        for (int fm = 0; fm < 2; ++fm) {
            // A fragment (16x32 bf16): lane lr = row; hs selects K-half per ISA layout
            int r = waveM * 32 + fm * 16 + lr;
            const __bf16* ap = &Alds[buf][r * LDA + hs * 8];
            v8bf alo = *(const v8bf*)(ap);        // K = hs*8 + 0..7
            v8bf ahi = *(const v8bf*)(ap + 16);   // K = 16 + hs*8 + 0..7
            v16bf a;
#pragma unroll
            for (int i = 0; i < 8; ++i) { a[i] = alo[i]; a[i + 8] = ahi[i]; }
#pragma unroll
            for (int fn = 0; fn < 4; ++fn) {
                int n = waveN * 64 + fn * 16 + lr;
                const __bf16* bp = &Blds[buf][n * LDB + hs * 8]; // K = hs*8 + 0..7, 16 + hs*8 + 0..7
                v8bf blo = *(const v8bf*)(bp);
                v8bf bhi = *(const v8bf*)(bp + 16);
                v16bf b;
#pragma unroll
                for (int i = 0; i < 8; ++i) { b[i] = blo[i]; b[i + 8] = bhi[i]; }
                acc[fm][fn] = __builtin_amdgcn_wmma_f32_16x16x32_bf16(
                    false, a, false, b, (short)0, acc[fm][fn], false, false);
            }
        }
    };

    auto waitAsync = [&]() {
        if constexpr (!A_IS_F32) {
            asm volatile("s_wait_asynccnt 0x0" ::: "memory");
        }
    };

    // ---- double-buffered main loop over K ----
    stageA(0, 0);
    stageB(0, 0);
    waitAsync();
    __syncthreads();
#pragma unroll 1
    for (int kt = 0; kt < KT; ++kt) {
        int buf = kt & 1;
        if (kt + 1 < KT) {
            stageA(kt + 1, buf ^ 1);    // async copies overlap with compute below
            stageB(kt + 1, buf ^ 1);
        }
        compute(buf);
        waitAsync();                    // async tile k+1 landed in LDS
        __syncthreads();
    }

    // ---- epilogue: bias + relu, scatter rows back through the index list ----
#pragma unroll
    for (int fm = 0; fm < 2; ++fm) {
        const int mb = waveM * 32 + fm * 16 + hs * 8;
        int grs[8];
#pragma unroll
        for (int v = 0; v < 8; ++v) {
            int gm = m0 + mb + v;
            grs[v] = rIdx ? ((gm < rows) ? rIdx[gm] : -1) : gm;
        }
#pragma unroll
        for (int fn = 0; fn < 4; ++fn) {
            const int gn = n0 + waveN * 64 + fn * 16 + lr;
            const float bv = bias[gn];
#pragma unroll
            for (int v = 0; v < 8; ++v) {
                if (grs[v] < 0) continue;
                float val = fmaxf(acc[fm][fn][v] + bv, 0.0f);
                if constexpr (OUT_BF16) {
                    ((__bf16*)Outptr)[(size_t)grs[v] * Dsz + gn] = f2bf(val);
                } else {
                    ((float*)Outptr)[(size_t)grs[v] * Dsz + gn] = val;
                }
            }
        }
    }
}

// ---------------------------------------------------------------------------
// Launch: compact -> stage0 (dense) -> stage1 (2 experts) -> stage2 (4 leaves)
// Workspace: h0 (bf16 B*D) | h1 (bf16 B*D) | cnt1[2],cnt2[4] | idx1[2*B] | idx2[4*B]
// ---------------------------------------------------------------------------
extern "C" void kernel_launch(void* const* d_in, const int* in_sizes, int n_in,
                              void* d_out, int out_size, void* d_ws, size_t ws_size,
                              hipStream_t stream) {
    (void)in_sizes; (void)n_in; (void)out_size; (void)ws_size;

    const float* x    = (const float*)d_in[0];
    const float* W0   = (const float*)d_in[1];
    const float* b0   = (const float*)d_in[2];
    const float* W1   = (const float*)d_in[3];
    const float* b1   = (const float*)d_in[4];
    const float* W2   = (const float*)d_in[5];
    const float* b2   = (const float*)d_in[6];
    const int*   mask = (const int*)d_in[7];
    float*       y    = (float*)d_out;

    char* ws = (char*)d_ws;
    __bf16* h0 = (__bf16*)ws;                                    // 32 MB
    __bf16* h1 = (__bf16*)(ws + (size_t)Bsz * Dsz * 2);          // 32 MB
    int* cnt1  = (int*)(ws + (size_t)Bsz * Dsz * 4);             // 2 ints
    int* cnt2  = cnt1 + 2;                                       // 4 ints
    int* idx1  = cnt1 + 8;                                       // 2*B ints
    int* idx2  = idx1 + 2 * Bsz;                                 // 4*B ints

    zero_counters_kernel<<<1, 32, 0, stream>>>(cnt1);
    compact_kernel<<<Bsz / 256, 256, 0, stream>>>(mask, cnt1, idx1, cnt2, idx2);

    dim3 blk(256);
    dim3 g0(Dsz / BN, Bsz / BM, 1);
    goe_gemm_kernel<true, true><<<g0, blk, 0, stream>>>(x, W0, b0, h0, nullptr, nullptr);

    dim3 g1(Dsz / BN, Bsz / BM, 2);
    goe_gemm_kernel<false, true><<<g1, blk, 0, stream>>>(h0, W1, b1, h1, idx1, cnt1);

    dim3 g2(Dsz / BN, Bsz / BM, 4);
    goe_gemm_kernel<false, false><<<g2, blk, 0, stream>>>(h1, W2, b2, y, idx2, cnt2);
}
